// ResidualDepthwiseConvAdapter_90580860273018
// MI455X (gfx1250) — matrix operand order repaired and checked by reading, not compiled, not run on hardware
//
#include <hip/hip_runtime.h>
#include <hip/hip_bf16.h>

typedef __attribute__((ext_vector_type(16))) __bf16 v16bf;
typedef __attribute__((ext_vector_type(8)))  __bf16 v8bf;
typedef __attribute__((ext_vector_type(8)))  float  v8f;

union AFrag { v16bf v; struct { v8bf lo, hi; } p; };

__device__ __forceinline__ unsigned short f2bf(float x) {
  union { float f; unsigned u; } a; a.f = x;
  unsigned r = a.u + 0x7fffu + ((a.u >> 16) & 1u);   // round-to-nearest-even
  return (unsigned short)(r >> 16);
}

__device__ __forceinline__ v16bf load_a16(const unsigned short* p) {
  // A-matrix fragment: elems 0..7 at p[0..7], elems 8..15 at p[16..23] (ISA 16-bit A 16x32 layout)
  AFrag a;
  a.p.lo = *(const v8bf*)(p);
  a.p.hi = *(const v8bf*)(p + 16);
  return a.v;
}

// ---------------- K0: weights -> bf16 (w_red2 reordered to [o, tap, c], c fastest) ----------
__global__ void __launch_bounds__(256) k_prep(const float* __restrict__ w_red1,
                                              const float* __restrict__ w_red2,
                                              unsigned short* __restrict__ w1bf,
                                              unsigned short* __restrict__ w2bf) {
  int idx = blockIdx.x * 256 + threadIdx.x;
  if (idx < 32 * 512) w1bf[idx] = f2bf(w_red1[idx]);
  int i2 = idx - 32 * 512;
  if (i2 >= 0 && i2 < 32 * 6912) {
    int o = i2 / 6912, r = i2 % 6912;
    int tap = r >> 8, c = r & 255;                    // K order: (tap major, channel fast)
    w2bf[i2] = f2bf(w_red2[(o * 256 + c) * 27 + tap]);
  }
}

// ---------------- K1: per-sample 1x1 convs -> dynamic depthwise kernels --------------------
// X: [16, K, 27]   W: [O, K]   out: [16, O, 27]
__global__ void __launch_bounds__(256) k_feat(const float* __restrict__ X,
                                              const float* __restrict__ W,
                                              float* __restrict__ out, int O, int K) {
  int idx = blockIdx.x * 256 + threadIdx.x;
  int total = 16 * O * 27;
  if (idx >= total) return;
  int b = idx / (O * 27);
  int r = idx - b * O * 27;
  int o = r / 27;
  int s = r - o * 27;
  const float* xp = X + (size_t)b * K * 27 + s;
  const float* wp = W + (size_t)o * K;
  float acc = 0.f;
  for (int k = 0; k < K; k += 4) {
    acc += wp[k + 0] * xp[(k + 0) * 27];
    acc += wp[k + 1] * xp[(k + 1) * 27];
    acc += wp[k + 2] * xp[(k + 2) * 27];
    acc += wp[k + 3] * xp[(k + 3) * 27];
  }
  out[idx] = acc;
}

// ---------------- K2: f2 dynamic depthwise (stride1, pad1) + ReLU -> act1 bf16 [16,4096,512]
__global__ void __launch_bounds__(256) k_dw_f2(const float* __restrict__ f2,
                                               const float* __restrict__ feat1,
                                               const float* __restrict__ feat2,
                                               unsigned short* __restrict__ act1) {
  int z = blockIdx.x;      // 16
  int b = blockIdx.y;      // 16
  int t = threadIdx.x;     // 256 input channels; computes output channels t and t+256
  const float* xin = f2 + ((size_t)(b * 256 + t)) * 4096;
  const float* p1 = feat1 + (size_t)(b * 256 + t) * 27;
  const float* p2 = feat2 + (size_t)(b * 256 + t) * 27;
  float k1[27], k2[27];
#pragma unroll
  for (int i = 0; i < 27; ++i) { k1[i] = p1[i]; k2[i] = p2[i]; }
  for (int y = 0; y < 16; ++y) {
    for (int x = 0; x < 16; ++x) {
      float a1 = 0.f, a2 = 0.f;
#pragma unroll
      for (int kd = 0; kd < 3; ++kd) {
        int zz = z - 1 + kd; if (zz < 0 || zz >= 16) continue;
#pragma unroll
        for (int kh = 0; kh < 3; ++kh) {
          int yy = y - 1 + kh; if (yy < 0 || yy >= 16) continue;
          int base = (zz * 16 + yy) * 16;
#pragma unroll
          for (int kw = 0; kw < 3; ++kw) {
            int xx = x - 1 + kw; if (xx < 0 || xx >= 16) continue;
            float v = xin[base + xx];
            int ki = (kd * 3 + kh) * 3 + kw;
            a1 += k1[ki] * v; a2 += k2[ki] * v;
          }
        }
      }
      size_t o = ((size_t)(b * 4096 + (z * 16 + y) * 16 + x)) << 9;  // *512 ch
      act1[o + t]       = f2bf(fmaxf(a1, 0.f));      // coalesced along t
      act1[o + 256 + t] = f2bf(fmaxf(a2, 0.f));
    }
  }
}

// ---------------- K3: out1 GEMM via bf16 WMMA: y[:,0:32,:] = w_red1 x act1 -----------------
// act1: [16, 4096, 512] bf16 (channel-last)   w1bf: [32,512]   ybuf: [16,64,4096] f32
__global__ void __launch_bounds__(256) k_gemm_out1(const unsigned short* __restrict__ act1,
                                                   const unsigned short* __restrict__ w1bf,
                                                   float* __restrict__ ybuf) {
  int wid  = (blockIdx.x * 256 + threadIdx.x) >> 5;
  int lane = threadIdx.x & 31;
  int b    = wid >> 8;                 // 16 batches
  int tile = wid & 255;                // 256 N-tiles of 16 spatial points
  int sbase = tile << 4;
  int h  = lane >> 4;
  int mn = lane & 15;

  const unsigned short* bp = act1 + (((size_t)(b * 4096 + sbase + mn)) << 9) + (h << 3);
  const unsigned short* a0 = w1bf + (size_t)mn * 512 + 8 * h;
  const unsigned short* a1 = w1bf + (size_t)(16 + mn) * 512 + 8 * h;

  v8f acc0 = {}; v8f acc1 = {};
#pragma unroll 4
  for (int k = 0; k < 512; k += 32) {
    v16bf bfrag = load_a16(bp + k);                  // two 16B loads per lane (table k order)
    v16bf af0 = load_a16(a0 + k);
    v16bf af1 = load_a16(a1 + k);
    acc0 = __builtin_amdgcn_wmma_f32_16x16x32_bf16(false, af0, false, bfrag, (short)0, acc0, false, false);
    acc1 = __builtin_amdgcn_wmma_f32_16x16x32_bf16(false, af1, false, bfrag, (short)0, acc1, false, false);
  }
  float* yb = ybuf + (size_t)b * 64 * 4096 + sbase + mn;
#pragma unroll
  for (int i = 0; i < 8; ++i) {
    yb[(size_t)(i + 8 * h) * 4096]      = acc0[i];   // rows o = i+8h
    yb[(size_t)(16 + i + 8 * h) * 4096] = acc1[i];   // rows o = 16+i+8h
  }
}

// ---------------- K4: f1 dynamic depthwise (stride2, pad1) + ReLU -> act2 bf16 [16,5832,256]
__global__ void __launch_bounds__(128) k_dw_f1(const float* __restrict__ f1,
                                               const float* __restrict__ feat1h,
                                               const float* __restrict__ feat2h,
                                               unsigned short* __restrict__ act2) {
  int z = blockIdx.x;      // 18
  int b = blockIdx.y;      // 16
  int t = threadIdx.x;     // 128 input channels; outputs t and t+128
  const float* xin = f1 + ((size_t)(b * 128 + t)) * 46656;  // 36^3
  const float* p1 = feat1h + (size_t)(b * 128 + t) * 27;
  const float* p2 = feat2h + (size_t)(b * 128 + t) * 27;
  float k1[27], k2[27];
#pragma unroll
  for (int i = 0; i < 27; ++i) { k1[i] = p1[i]; k2[i] = p2[i]; }
  for (int y = 0; y < 18; ++y) {
    for (int x = 0; x < 18; ++x) {
      float a1 = 0.f, a2 = 0.f;
#pragma unroll
      for (int kd = 0; kd < 3; ++kd) {
        int zz = 2 * z - 1 + kd; if (zz < 0 || zz >= 36) continue;
#pragma unroll
        for (int kh = 0; kh < 3; ++kh) {
          int yy = 2 * y - 1 + kh; if (yy < 0 || yy >= 36) continue;
          int base = (zz * 36 + yy) * 36;
#pragma unroll
          for (int kw = 0; kw < 3; ++kw) {
            int xx = 2 * x - 1 + kw; if (xx < 0 || xx >= 36) continue;
            float v = xin[base + xx];
            int ki = (kd * 3 + kh) * 3 + kw;
            a1 += k1[ki] * v; a2 += k2[ki] * v;
          }
        }
      }
      size_t o = ((size_t)(b * 5832 + (z * 18 + y) * 18 + x)) << 8;  // *256 ch
      act2[o + t]       = f2bf(fmaxf(a1, 0.f));
      act2[o + 128 + t] = f2bf(fmaxf(a2, 0.f));
    }
  }
}

// ---------------- K5: out2 implicit GEMM (3^3 VALID conv) via bf16 WMMA -> y[:,32:64,:] ----
// act2: [16, 18^3, 256] bf16   w2bf: [32, 27*256]   K = 6912, N-tile = x-strip of 16
__global__ void __launch_bounds__(256) k_gemm_out2(const unsigned short* __restrict__ act2,
                                                   const unsigned short* __restrict__ w2bf,
                                                   float* __restrict__ ybuf) {
  int wid  = (blockIdx.x * 256 + threadIdx.x) >> 5;
  int lane = threadIdx.x & 31;
  int b  = wid >> 8;                   // 16
  int zy = wid & 255;                  // (z,y) of out2, 16x16
  int z = zy >> 4, y = zy & 15;
  int h  = lane >> 4;
  int mn = lane & 15;                  // B col = out x = mn

  const unsigned short* xb = act2 + (size_t)b * 5832 * 256 + (h << 3);
  const unsigned short* a0 = w2bf + (size_t)mn * 6912 + 8 * h;
  const unsigned short* a1 = w2bf + (size_t)(16 + mn) * 6912 + 8 * h;

  v8f acc0 = {}; v8f acc1 = {};
  for (int tap = 0; tap < 27; ++tap) {
    int kd = tap / 9, rr = tap - kd * 9, kh = rr / 3, kw = rr - kh * 3;
    const unsigned short* brow =
        xb + (((size_t)(((z + kd) * 18 + (y + kh)) * 18 + (mn + kw))) << 8);
    int ak = tap << 8;
#pragma unroll
    for (int cc = 0; cc < 8; ++cc) {
      int cb = cc << 5;
      v16bf bfrag = load_a16(brow + cb);
      v16bf af0 = load_a16(a0 + ak + cb);
      v16bf af1 = load_a16(a1 + ak + cb);
      acc0 = __builtin_amdgcn_wmma_f32_16x16x32_bf16(false, af0, false, bfrag, (short)0, acc0, false, false);
      acc1 = __builtin_amdgcn_wmma_f32_16x16x32_bf16(false, af1, false, bfrag, (short)0, acc1, false, false);
    }
  }
  int s = zy * 16 + mn;                                // (z*16+y)*16 + x
  float* yb = ybuf + (size_t)b * 64 * 4096 + (size_t)32 * 4096 + s;
#pragma unroll
  for (int i = 0; i < 8; ++i) {
    yb[(size_t)(i + 8 * h) * 4096]      = acc0[i];
    yb[(size_t)(16 + i + 8 * h) * 4096] = acc1[i];
  }
}

// ---------------- K6: global mean pool + gate sigmoid ------------------------------------
__global__ void __launch_bounds__(256) k_gate(const float* __restrict__ ybuf,
                                              const float* __restrict__ w_gate,
                                              const float* __restrict__ b_gate,
                                              float* __restrict__ scale) {
  __shared__ float red[256];
  __shared__ float pooled[64];
  int b = blockIdx.x, t = threadIdx.x;
  for (int ch = 0; ch < 64; ++ch) {
    const float* yp = ybuf + ((size_t)(b * 64 + ch)) * 4096;
    float p = 0.f;
    for (int i = t; i < 4096; i += 256) p += yp[i];
    red[t] = p; __syncthreads();
    for (int off = 128; off > 0; off >>= 1) {
      if (t < off) red[t] += red[t + off];
      __syncthreads();
    }
    if (t == 0) pooled[ch] = red[0] * (1.f / 4096.f);
    __syncthreads();
  }
  if (t < 16) {
    float g = b_gate[t];
    for (int c = 0; c < 64; ++c) g += w_gate[t * 64 + c] * pooled[c];
    scale[b * 16 + t] = 1.f / (1.f + __expf(-g));
  }
}

// ---------------- K7: fuse conv (stride2 VALID 3^3, 64->16 ch) + bias -> z2 [16,16,343] ----
__global__ void __launch_bounds__(256) k_fuse(const float* __restrict__ ybuf,
                                              const float* __restrict__ w_fuse,
                                              const float* __restrict__ b_fuse,
                                              float* __restrict__ z2) {
  int idx = blockIdx.x * 256 + threadIdx.x;           // 16*16*343 = 87808 exact
  int b = idx / (16 * 343);
  int r = idx - b * 16 * 343;
  int oc = r / 343;
  int s = r - oc * 343;
  int zo = s / 49, r2 = s - zo * 49, yo = r2 / 7, xo = r2 - yo * 7;
  float acc = b_fuse[oc];
  const float* wb = w_fuse + (size_t)oc * 64 * 27;
  for (int c = 0; c < 64; ++c) {
    const float* yp = ybuf + ((size_t)(b * 64 + c)) * 4096;
    const float* wp = wb + c * 27;
#pragma unroll
    for (int kd = 0; kd < 3; ++kd)
#pragma unroll
      for (int kh = 0; kh < 3; ++kh)
#pragma unroll
        for (int kw = 0; kw < 3; ++kw)
          acc += yp[((2 * zo + kd) * 16 + (2 * yo + kh)) * 16 + (2 * xo + kw)] *
                 wp[(kd * 3 + kh) * 3 + kw];
  }
  z2[idx] = acc;
}

// ---------------- K8: batch-stats per channel (N = 16*343) --------------------------------
__global__ void __launch_bounds__(256) k_bnstats(const float* __restrict__ z2,
                                                 float* __restrict__ bn) {
  __shared__ float rs[256], rq[256];
  int ch = blockIdx.x, t = threadIdx.x;
  float s = 0.f, q = 0.f;
  for (int i = t; i < 16 * 343; i += 256) {
    int b = i / 343, sp = i - b * 343;
    float v = z2[(size_t)(b * 16 + ch) * 343 + sp];
    s += v; q += v * v;
  }
  rs[t] = s; rq[t] = q; __syncthreads();
  for (int off = 128; off > 0; off >>= 1) {
    if (t < off) { rs[t] += rs[t + off]; rq[t] += rq[t + off]; }
    __syncthreads();
  }
  if (t == 0) {
    float mean = rs[0] * (1.f / 5488.f);
    float var  = rq[0] * (1.f / 5488.f) - mean * mean;
    bn[ch]      = mean;
    bn[16 + ch] = rsqrtf(var + 1e-5f);
  }
}

// ---------------- K9: normalize, affine, ReLU, gate-scale, flatten ------------------------
__global__ void __launch_bounds__(256) k_final(const float* __restrict__ z2,
                                               const float* __restrict__ bn,
                                               const float* __restrict__ scale,
                                               const float* __restrict__ gamma,
                                               const float* __restrict__ beta,
                                               float* __restrict__ out) {
  int idx = blockIdx.x * 256 + threadIdx.x;           // 16*5488 = 87808 exact
  int b = idx / 5488;
  int r = idx - b * 5488;
  int ch = r / 343;
  float v = z2[idx];                                  // z2 flat layout == output layout
  v = (v - bn[ch]) * bn[16 + ch] * gamma[ch] + beta[ch];
  v = fmaxf(v, 0.f) * scale[b * 16 + ch];
  out[idx] = v;
}

extern "C" void kernel_launch(void* const* d_in, const int* in_sizes, int n_in,
                              void* d_out, int out_size, void* d_ws, size_t ws_size,
                              hipStream_t stream) {
  (void)in_sizes; (void)n_in; (void)out_size; (void)ws_size;
  const float* f1        = (const float*)d_in[0];
  const float* f2        = (const float*)d_in[1];
  const float* f3        = (const float*)d_in[2];
  const float* f4        = (const float*)d_in[3];
  const float* w_conv1   = (const float*)d_in[4];
  const float* w_reduce1 = (const float*)d_in[5];
  const float* w_conv2   = (const float*)d_in[6];
  const float* w_reduce2 = (const float*)d_in[7];
  const float* w_red1    = (const float*)d_in[8];
  const float* w_red2    = (const float*)d_in[9];
  const float* w_gate    = (const float*)d_in[10];
  const float* b_gate    = (const float*)d_in[11];
  const float* w_fuse    = (const float*)d_in[12];
  const float* b_fuse    = (const float*)d_in[13];
  const float* gamma     = (const float*)d_in[14];
  const float* beta      = (const float*)d_in[15];

  char* ws = (char*)d_ws;
  size_t off = 0;
  auto alloc = [&](size_t bytes) -> char* {
    char* p = ws + off;
    off += (bytes + 255) & ~(size_t)255;
    return p;
  };
  float* feat1  = (float*)alloc((size_t)16 * 256 * 27 * 4);
  float* feat1h = (float*)alloc((size_t)16 * 128 * 27 * 4);
  float* feat2  = (float*)alloc((size_t)16 * 256 * 27 * 4);
  float* feat2h = (float*)alloc((size_t)16 * 128 * 27 * 4);
  unsigned short* w1bf = (unsigned short*)alloc((size_t)32 * 512 * 2);
  unsigned short* w2bf = (unsigned short*)alloc((size_t)32 * 6912 * 2);
  unsigned short* act1 = (unsigned short*)alloc((size_t)16 * 4096 * 512 * 2);  // 64 MB
  unsigned short* act2 = (unsigned short*)alloc((size_t)16 * 5832 * 256 * 2);  // 46 MB
  float* ybuf  = (float*)alloc((size_t)16 * 64 * 4096 * 4);                    // 16 MB
  float* scale = (float*)alloc((size_t)16 * 16 * 4);
  float* z2    = (float*)alloc((size_t)16 * 16 * 343 * 4);
  float* bn    = (float*)alloc((size_t)32 * 4);

  k_prep<<<928, 256, 0, stream>>>(w_red1, w_red2, w1bf, w2bf);
  k_feat<<<(16 * 256 * 27 + 255) / 256, 256, 0, stream>>>(f3, w_conv1,   feat1,  256, 512);
  k_feat<<<(16 * 128 * 27 + 255) / 256, 256, 0, stream>>>(f3, w_reduce1, feat1h, 128, 512);
  k_feat<<<(16 * 256 * 27 + 255) / 256, 256, 0, stream>>>(f4, w_conv2,   feat2,  256, 1024);
  k_feat<<<(16 * 128 * 27 + 255) / 256, 256, 0, stream>>>(f4, w_reduce2, feat2h, 128, 1024);
  k_dw_f2<<<dim3(16, 16), 256, 0, stream>>>(f2, feat1, feat2, act1);
  k_gemm_out1<<<512, 256, 0, stream>>>(act1, w1bf, ybuf);
  k_dw_f1<<<dim3(18, 16), 128, 0, stream>>>(f1, feat1h, feat2h, act2);
  k_gemm_out2<<<512, 256, 0, stream>>>(act2, w2bf, ybuf);
  k_gate<<<16, 256, 0, stream>>>(ybuf, w_gate, b_gate, scale);
  k_fuse<<<343, 256, 0, stream>>>(ybuf, w_fuse, b_fuse, z2);
  k_bnstats<<<16, 256, 0, stream>>>(z2, bn);
  k_final<<<343, 256, 0, stream>>>(z2, bn, scale, gamma, beta, (float*)d_out);
}
